// AlignAttendPooling_64046552318009
// MI455X (gfx1250) — matrix operand order repaired and checked by reading, not compiled, not run on hardware
//
#include <hip/hip_runtime.h>

// ---------------------------------------------------------------------------
// AlignAttendPooling for MI455X (gfx1250, wave32, WMMA).
//
// Structure exploited: mol_node_matrix is a one-hot assignment => segment
// ops + per-segment softmax instead of dense [1024 x 50000] GEMMs.
// Big N-dim GEMMs (mapped, h) run on V_WMMA_F32_16X16X32_BF16 (bf16 in,
// f32 accumulate) -> memory-bound instead of compute-bound.
// Weights are pre-converted to bf16 ONCE so the WMMA k-loop is pure
// loads + wmma (no f32->bf16 VALU rounding in the hot loop).
// Everything else (softmax, ELU, GRU, biases) is exact f32.
// ---------------------------------------------------------------------------

#define N_NODE 50000
#define N_MOL  1024
#define C_DIM  128
#define P_DIM  64

typedef __attribute__((ext_vector_type(16))) __bf16    v16bf;
typedef __attribute__((ext_vector_type(8)))  float     v8f;
typedef __attribute__((ext_vector_type(8)))  unsigned  v8u;

__device__ __forceinline__ unsigned short f2bf_u16(float f) {
  unsigned u = __builtin_bit_cast(unsigned, f);
  unsigned r = u + 0x7FFFu + ((u >> 16) & 1u);      // round-to-nearest-even
  return (unsigned short)(r >> 16);
}

// A-fragment K offset within a 32-wide K block (16-bit A 16x32 layout,
// ISA 7.12.2): lanes 0-15 -> VGPR p<4: K=2p,2p+1 ; p>=4: K=8+2p,9+2p.
// lanes 16-31: same +8.
__device__ __forceinline__ int a_koff(int p, int lane) {
  int k = (p < 4) ? (2 * p) : (8 + 2 * p);
  return k + ((lane >= 16) ? 8 : 0);
}

// Load A fragment: 8 dword loads of packed bf16 pairs from a row-major row.
__device__ __forceinline__ v16bf load_a_frag(const unsigned short* rowp, int lane) {
  v8u u;
#pragma unroll
  for (int p = 0; p < 8; ++p)
    u[p] = *(const unsigned*)(rowp + a_koff(p, lane));
  return __builtin_bit_cast(v16bf, u);
}

// B fragment (32x16, K x N): lane carries column n = lane&15; K order
// matches the A fragment (a_koff): VGPR p holds K=a_koff(p,lane),+1.
// Source is PRE-CONVERTED bf16 weight row Wbf[j][k..] -> 8 dword loads.
__device__ __forceinline__ v16bf load_b_frag(const unsigned short* wrow, int lane) {
  v8u u;
#pragma unroll
  for (int p = 0; p < 8; ++p)
    u[p] = *(const unsigned*)(wrow + a_koff(p, lane));
  return __builtin_bit_cast(v16bf, u);
}

// ---------------------------------------------------------------------------
// Phase 0: init / assignment / CSR
// ---------------------------------------------------------------------------

__global__ void zero_k(float* molf, int* counts) {
  int i = blockIdx.x * blockDim.x + threadIdx.x;
  if (i < N_MOL * C_DIM) molf[i] = 0.0f;
  if (i < N_MOL)         counts[i] = 0;
}

__global__ void assign_k(const float* __restrict__ mat, int* __restrict__ assign,
                         int* __restrict__ counts) {
  int n = blockIdx.x * blockDim.x + threadIdx.x;
  if (n >= N_NODE) return;
  int a = 0;
  for (int m = 0; m < N_MOL; ++m) {           // coalesced across threads
    if (mat[(size_t)m * N_NODE + n] > 0.5f) { a = m; break; }
  }
  assign[n] = a;
  atomicAdd(&counts[a], 1);
}

__global__ void scan_k(const int* __restrict__ counts, int* __restrict__ offsets,
                       int* __restrict__ cursor) {
  __shared__ int s[N_MOL];
  int t = threadIdx.x;
  int c = counts[t];
  s[t] = c;
  __syncthreads();
  for (int d = 1; d < N_MOL; d <<= 1) {
    int add = (t >= d) ? s[t - d] : 0;
    __syncthreads();
    s[t] += add;
    __syncthreads();
  }
  offsets[t] = s[t] - c;   // exclusive
  cursor[t]  = s[t] - c;
}

__global__ void fill_k(const int* __restrict__ assign, int* __restrict__ cursor,
                       int* __restrict__ node_list) {
  int n = blockIdx.x * blockDim.x + threadIdx.x;
  if (n >= N_NODE) return;
  int slot = atomicAdd(&cursor[assign[n]], 1);
  node_list[slot] = n;
}

__global__ void cvt_bf16_k(const float* __restrict__ src,
                           unsigned short* __restrict__ dst, int n) {
  int i = blockIdx.x * blockDim.x + threadIdx.x;
  if (i < n) dst[i] = f2bf_u16(src[i]);
}

// ---------------------------------------------------------------------------
// GEMM 1: molF0[m] += leaky(node @ map_W^T + map_b), pooled by assignment.
// Block = 4 waves = one 16-row tile x all 8 column tiles (2 per wave).
// K=128 -> 4 k-blocks x 2 x v_wmma_f32_16x16x32_bf16 per wave.
// ---------------------------------------------------------------------------
__global__ void __launch_bounds__(128) gemm_map_pool_k(
    const unsigned short* __restrict__ Abf,   // [N_NODE,128] bf16
    const unsigned short* __restrict__ Wbf,   // map_W [128,128] bf16
    const float* __restrict__ bias,           // map_b [128]
    const int* __restrict__ assign,
    float* __restrict__ molf)                 // [1024,128], pre-zeroed
{
  int lane = threadIdx.x & 31;
  int wv   = threadIdx.x >> 5;
  int rowTile = blockIdx.x;                   // N_NODE/16 tiles
  int row = rowTile * 16 + (lane & 15);
  int j0  = wv * 16 + (lane & 15);            // col tiles wv and wv+4
  int j1  = j0 + 64;

  v8f acc0 = {}, acc1 = {};
#pragma unroll
  for (int kb = 0; kb < 4; ++kb) {
    v16bf a  = load_a_frag(Abf + (size_t)row * C_DIM + kb * 32, lane);
    v16bf b0 = load_b_frag(Wbf + j0 * C_DIM + kb * 32, lane);
    acc0 = __builtin_amdgcn_wmma_f32_16x16x32_bf16(false, a, false, b0,
                                                   (short)0, acc0, false, false);
    v16bf b1 = load_b_frag(Wbf + j1 * C_DIM + kb * 32, lane);
    acc1 = __builtin_amdgcn_wmma_f32_16x16x32_bf16(false, a, false, b1,
                                                   (short)0, acc1, false, false);
  }
  float bj0 = bias[j0], bj1 = bias[j1];
  int rbase = rowTile * 16 + ((lane >= 16) ? 8 : 0);
#pragma unroll
  for (int r = 0; r < 8; ++r) {
    int mrow = assign[rbase + r] * C_DIM;
    float v0 = acc0[r] + bj0;
    v0 = (v0 > 0.0f) ? v0 : 0.01f * v0;       // leaky before pooling
    atomicAdd(&molf[mrow + j0], v0);
    float v1 = acc1[r] + bj1;
    v1 = (v1 > 0.0f) ? v1 : 0.01f * v1;
    atomicAdd(&molf[mrow + j1], v1);
  }
}

// ---------------------------------------------------------------------------
// GEMM 2: h = [node,pos] @ attend_W^T + attend_b  (loop-invariant, once)
// K=192 -> 6 k-blocks; A source switches node->pos at kb==4.
// ---------------------------------------------------------------------------
__global__ void __launch_bounds__(128) gemm_h_k(
    const unsigned short* __restrict__ nodebf, // [N,128] bf16
    const unsigned short* __restrict__ posbf,  // [N,64]  bf16
    const unsigned short* __restrict__ Wbf,    // attend_W [128,192] bf16
    const float* __restrict__ bias,            // attend_b
    float* __restrict__ h)                     // [N,128] f32
{
  int lane = threadIdx.x & 31;
  int wv   = threadIdx.x >> 5;
  int rowTile = blockIdx.x;
  int row = rowTile * 16 + (lane & 15);
  int j0  = wv * 16 + (lane & 15);
  int j1  = j0 + 64;

  v8f acc0 = {}, acc1 = {};
#pragma unroll
  for (int kb = 0; kb < 6; ++kb) {
    const unsigned short* ap = (kb < 4)
        ? (nodebf + (size_t)row * C_DIM + kb * 32)
        : (posbf  + (size_t)row * P_DIM + (kb - 4) * 32);
    v16bf a  = load_a_frag(ap, lane);
    v16bf b0 = load_b_frag(Wbf + j0 * 192 + kb * 32, lane);
    acc0 = __builtin_amdgcn_wmma_f32_16x16x32_bf16(false, a, false, b0,
                                                   (short)0, acc0, false, false);
    v16bf b1 = load_b_frag(Wbf + j1 * 192 + kb * 32, lane);
    acc1 = __builtin_amdgcn_wmma_f32_16x16x32_bf16(false, a, false, b1,
                                                   (short)0, acc1, false, false);
  }
  float bj0 = bias[j0], bj1 = bias[j1];
  int rbase = rowTile * 16 + ((lane >= 16) ? 8 : 0);
#pragma unroll
  for (int r = 0; r < 8; ++r) {
    h[(size_t)(rbase + r) * C_DIM + j0] = acc0[r] + bj0;
    h[(size_t)(rbase + r) * C_DIM + j1] = acc1[r] + bj1;
  }
}

// ---------------------------------------------------------------------------
// t[n] = align_W[128:192].pos[n] + align_W[192:320].node[n] + align_b
// (loop-invariant part of the align score). One wave per node.
// ---------------------------------------------------------------------------
__global__ void __launch_bounds__(128) tvec_k(
    const float* __restrict__ node, const float* __restrict__ pos,
    const float* __restrict__ alignW, const float* __restrict__ alignb,
    float* __restrict__ t)
{
  int lane = threadIdx.x & 31;
  int wv   = threadIdx.x >> 5;
  int n = blockIdx.x * 4 + wv;
  if (n >= N_NODE) return;
  const float* wp = alignW + 128;   // pos part
  const float* wn = alignW + 192;   // node part
  float s = 0.0f;
#pragma unroll
  for (int i = 0; i < 4; ++i) {
    int k = lane + 32 * i;
    s += wn[k] * node[(size_t)n * C_DIM + k];
  }
#pragma unroll
  for (int i = 0; i < 2; ++i) {
    int k = lane + 32 * i;
    s += wp[k] * pos[(size_t)n * P_DIM + k];
  }
#pragma unroll
  for (int d = 16; d >= 1; d >>= 1) s += __shfl_xor(s, d, 32);
  if (lane == 0) t[n] = s + alignb[0];
}

// s_mol[m] = molF[m] . align_W[0:128]
__global__ void smol_k(const float* __restrict__ molf,
                       const float* __restrict__ alignW,
                       float* __restrict__ smol) {
  int m = blockIdx.x * blockDim.x + threadIdx.x;
  if (m >= N_MOL) return;
  float s = 0.0f;
  for (int k = 0; k < C_DIM; ++k) s += molf[m * C_DIM + k] * alignW[k];
  smol[m] = s;
}

// ---------------------------------------------------------------------------
// Per-molecule masked softmax + weighted sum of h + ELU.
// Block = one molecule, thread = one channel; h reads are coalesced 512B rows.
// exp(-1e9 - amax) underflows to exactly 0 in f32, so the segment softmax is
// bit-equivalent to the masked dense softmax.
// ---------------------------------------------------------------------------
__global__ void __launch_bounds__(128) context_k(
    const float* __restrict__ h, const float* __restrict__ t,
    const float* __restrict__ smol,
    const int* __restrict__ offsets, const int* __restrict__ counts,
    const int* __restrict__ node_list,
    float* __restrict__ context)
{
  int m = blockIdx.x, c = threadIdx.x;
  int cnt = counts[m], off = offsets[m];
  float sm = smol[m];

  float amax = -3.0e38f;                       // redundant per-thread (broadcast loads)
  for (int i = 0; i < cnt; ++i) {
    float a = sm + t[node_list[off + i]];
    a = (a > 0.0f) ? a : 0.01f * a;            // leaky
    amax = fmaxf(amax, a);
  }
  float Z = 0.0f, acc = 0.0f;
  for (int i = 0; i < cnt; ++i) {
    int n = node_list[off + i];
    float a = sm + t[n];
    a = (a > 0.0f) ? a : 0.01f * a;
    float wgt = __expf(a - amax);
    Z += wgt;
    acc += wgt * h[(size_t)n * C_DIM + c];
  }
  float v = (cnt > 0) ? (acc / Z) : 0.0f;
  context[m * C_DIM + c] = (v > 0.0f) ? v : (__expf(v) - 1.0f);  // ELU
}

// ---------------------------------------------------------------------------
// GRU cell + ReLU, exact f32. Block = molecule, thread = output channel.
// State rows staged in LDS; weights (196 KB each) are L2-resident.
// ---------------------------------------------------------------------------
__global__ void __launch_bounds__(128) gru_k(
    const float* __restrict__ x,  const float* __restrict__ hprev,
    const float* __restrict__ Wih, const float* __restrict__ Whh,
    const float* __restrict__ bih, const float* __restrict__ bhh,
    float* __restrict__ out)
{
  int m = blockIdx.x, j = threadIdx.x;
  __shared__ float xs[C_DIM], hs[C_DIM];
  xs[j] = x[m * C_DIM + j];
  hs[j] = hprev[m * C_DIM + j];
  __syncthreads();

  float ir = bih[j], iz = bih[C_DIM + j], inn = bih[2 * C_DIM + j];
  float hr = bhh[j], hz = bhh[C_DIM + j], hn  = bhh[2 * C_DIM + j];
  const float* wi0 = Wih + (size_t)j * C_DIM;
  const float* wi1 = Wih + (size_t)(C_DIM + j) * C_DIM;
  const float* wi2 = Wih + (size_t)(2 * C_DIM + j) * C_DIM;
  const float* wh0 = Whh + (size_t)j * C_DIM;
  const float* wh1 = Whh + (size_t)(C_DIM + j) * C_DIM;
  const float* wh2 = Whh + (size_t)(2 * C_DIM + j) * C_DIM;

  for (int k = 0; k < C_DIM; k += 4) {
    float4 a0 = *(const float4*)(wi0 + k);
    float4 a1 = *(const float4*)(wi1 + k);
    float4 a2 = *(const float4*)(wi2 + k);
    float4 b0 = *(const float4*)(wh0 + k);
    float4 b1 = *(const float4*)(wh1 + k);
    float4 b2 = *(const float4*)(wh2 + k);
    float x0 = xs[k], x1 = xs[k + 1], x2 = xs[k + 2], x3 = xs[k + 3];
    float h0 = hs[k], h1 = hs[k + 1], h2 = hs[k + 2], h3 = hs[k + 3];
    ir  += a0.x * x0 + a0.y * x1 + a0.z * x2 + a0.w * x3;
    iz  += a1.x * x0 + a1.y * x1 + a1.z * x2 + a1.w * x3;
    inn += a2.x * x0 + a2.y * x1 + a2.z * x2 + a2.w * x3;
    hr  += b0.x * h0 + b0.y * h1 + b0.z * h2 + b0.w * h3;
    hz  += b1.x * h0 + b1.y * h1 + b1.z * h2 + b1.w * h3;
    hn  += b2.x * h0 + b2.y * h1 + b2.z * h2 + b2.w * h3;
  }
  float r = 1.0f / (1.0f + __expf(-(ir + hr)));
  float z = 1.0f / (1.0f + __expf(-(iz + hz)));
  float nn = tanhf(inn + r * hn);
  float o = (1.0f - z) * nn + z * hs[j];
  out[m * C_DIM + j] = fmaxf(o, 0.0f);
}

// ---------------------------------------------------------------------------
// Host launcher
// ---------------------------------------------------------------------------
extern "C" void kernel_launch(void* const* d_in, const int* in_sizes, int n_in,
                              void* d_out, int out_size, void* d_ws, size_t ws_size,
                              hipStream_t stream) {
  const float* node   = (const float*)d_in[0];
  const float* pos    = (const float*)d_in[1];
  const float* mat    = (const float*)d_in[2];
  /* d_in[3] = mol_node_mask: implied by the one-hot matrix, unused */
  const float* mapW   = (const float*)d_in[4];
  const float* mapb   = (const float*)d_in[5];
  const float* attW   = (const float*)d_in[6];
  const float* attb   = (const float*)d_in[7];
  const float* alignW = (const float*)d_in[8];
  const float* alignb = (const float*)d_in[9];
  const float* Wih    = (const float*)d_in[10];
  const float* Whh    = (const float*)d_in[11];
  const float* bih    = (const float*)d_in[12];
  const float* bhh    = (const float*)d_in[13];
  float* out = (float*)d_out;

  // workspace carve-out (~47 MB total)
  char* w = (char*)d_ws;
  size_t o = 0;
  auto alloc = [&](size_t bytes) -> char* {
    char* p = w + o;
    o += (bytes + 255) & ~(size_t)255;
    return p;
  };
  float* molfA     = (float*)alloc(N_MOL * C_DIM * 4);
  float* molfB     = (float*)alloc(N_MOL * C_DIM * 4);
  float* smol      = (float*)alloc(N_MOL * 4);
  float* context   = (float*)alloc(N_MOL * C_DIM * 4);
  float* tvec      = (float*)alloc((size_t)N_NODE * 4);
  int*   assign    = (int*)  alloc((size_t)N_NODE * 4);
  int*   counts    = (int*)  alloc(N_MOL * 4);
  int*   offsets   = (int*)  alloc(N_MOL * 4);
  int*   cursor    = (int*)  alloc(N_MOL * 4);
  int*   node_list = (int*)  alloc((size_t)N_NODE * 4);
  unsigned short* nodebf = (unsigned short*)alloc((size_t)N_NODE * C_DIM * 2);
  unsigned short* posbf  = (unsigned short*)alloc((size_t)N_NODE * P_DIM * 2);
  unsigned short* mapWbf = (unsigned short*)alloc((size_t)C_DIM * C_DIM * 2);
  unsigned short* attWbf = (unsigned short*)alloc((size_t)C_DIM * 192 * 2);
  float* hbuf      = (float*)alloc((size_t)N_NODE * C_DIM * 4);
  (void)ws_size; (void)in_sizes; (void)n_in; (void)out_size;

  // Phase 0: init, assignment, CSR, bf16 conversions (features AND weights)
  zero_k<<<(N_MOL * C_DIM + 255) / 256, 256, 0, stream>>>(molfA, counts);
  assign_k<<<(N_NODE + 255) / 256, 256, 0, stream>>>(mat, assign, counts);
  scan_k<<<1, N_MOL, 0, stream>>>(counts, offsets, cursor);
  fill_k<<<(N_NODE + 255) / 256, 256, 0, stream>>>(assign, cursor, node_list);
  cvt_bf16_k<<<((size_t)N_NODE * C_DIM + 255) / 256, 256, 0, stream>>>(node, nodebf, N_NODE * C_DIM);
  cvt_bf16_k<<<((size_t)N_NODE * P_DIM + 255) / 256, 256, 0, stream>>>(pos, posbf, N_NODE * P_DIM);
  cvt_bf16_k<<<(C_DIM * C_DIM + 255) / 256, 256, 0, stream>>>(mapW, mapWbf, C_DIM * C_DIM);
  cvt_bf16_k<<<(C_DIM * 192 + 255) / 256, 256, 0, stream>>>(attW, attWbf, C_DIM * 192);

  // Phase 1: loop-invariant heavy lifting (WMMA GEMMs)
  gemm_map_pool_k<<<N_NODE / 16, 128, 0, stream>>>(nodebf, mapWbf, mapb, assign, molfA);
  tvec_k<<<N_NODE / 4, 128, 0, stream>>>(node, pos, alignW, alignb, tvec);
  gemm_h_k<<<N_NODE / 16, 128, 0, stream>>>(nodebf, posbf, attWbf, attb, hbuf);

  // Phase 2: RADIUS = 2 iterations (h, t are invariant; only molF evolves)
  smol_k<<<N_MOL / 128, 128, 0, stream>>>(molfA, alignW, smol);
  context_k<<<N_MOL, 128, 0, stream>>>(hbuf, tvec, smol, offsets, counts, node_list, context);
  gru_k<<<N_MOL, 128, 0, stream>>>(context, molfA, Wih, Whh, bih, bhh, molfB);

  smol_k<<<N_MOL / 128, 128, 0, stream>>>(molfB, alignW, smol);
  context_k<<<N_MOL, 128, 0, stream>>>(hbuf, tvec, smol, offsets, counts, node_list, context);
  gru_k<<<N_MOL, 128, 0, stream>>>(context, molfB, Wih, Whh, bih, bhh, out);
}
